// PGNetBeamSerachLayer_68238440398931
// MI455X (gfx1250) — matrix operand order repaired and checked by reading, not compiled, not run on hardware
//
#include <hip/hip_runtime.h>

// ---------------- constants (match reference) ----------------
constexpr int NB    = 64;      // batch
constexpr int SLEN  = 400;     // encoder length
constexpr int HD    = 256;     // hidden
constexpr int ED    = 128;     // embedding dim
constexpr int NV    = 50000;   // vocab
constexpr int NOOV  = 100;     // max oov
constexpr int NVE   = NV + NOOV;
constexpr int NBEAM = 4;
constexpr int NSTEP = 32;
constexpr int NTIL  = NV / 16; // 3125 WMMA column tiles
constexpr int PANEL = 16 * HD; // 4096 bf16 elements per staged B panel (8KB)

typedef float  v8f   __attribute__((ext_vector_type(8)));
typedef __bf16 v16bf __attribute__((ext_vector_type(16)));
typedef unsigned int u32x4 __attribute__((ext_vector_type(4)));
typedef int          i32x4 __attribute__((ext_vector_type(4)));
typedef int          i32x8 __attribute__((ext_vector_type(8)));

// ---------------- shared memory plan ----------------
struct Smem {
  float att[NBEAM * SLEN];                 // attention scores -> weights
  float h[NBEAM * HD], c[NBEAM * HD], ctx[NBEAM * HD];
  float hN[NBEAM * HD], cN[NBEAM * HD], ctxN[NBEAM * HD];
  float emb[NBEAM * ED], xbuf[NBEAM * ED];
  float z[NBEAM * 4 * HD];
  float df[NBEAM * HD], out1[NBEAM * HD];
  unsigned short Apad[16 * HD];            // bf16 padded A matrix (16 x 256)
  float vatt[HD], batt[HD];
  float red[256];
  float tkv[256 * 4]; int tki[256 * 4];
  float wavemax[8 * NBEAM];
  int   tokens[NBEAM * NSTEP], tokensN[NBEAM * NSTEP];
  float lp[NBEAM], vals[NBEAM], pg[NBEAM], sums[NBEAM], mx[NBEAM];
  int   tokc[NBEAM], ntok[NBEAM], bsel[NBEAM];
  // TDM double-buffered B panels: 8 waves x 2 buffers x 4096 bf16 = 128KB
  unsigned short bstage[8 * 2 * PANEL] __attribute__((aligned(32)));
};

__device__ inline float sigm(float x) { return 1.f / (1.f + __expf(-x)); }

__device__ inline float wredsum(float v) {
  #pragma unroll
  for (int m = 16; m; m >>= 1) v += __shfl_xor(v, m, 32);
  return v;
}
__device__ inline float wredmax(float v) {
  #pragma unroll
  for (int m = 16; m; m >>= 1) v = fmaxf(v, __shfl_xor(v, m, 32));
  return v;
}

// ---- Tensor Data Mover: async copy of one contiguous 8KB B panel to LDS ----
// D# per CDNA5 ISA: group0 = {count|flags, lds_addr,
// global_addr[31:0], global_addr[56:32]|type=2}; group1 carries data_size,
// tensor_dim0, tile_dim0 (1-D tile: tile_dim1 = 0).
__device__ inline void tdm_load_panel(const unsigned short* Wt, int n0,
                                      unsigned ldsoff) {
  unsigned long long ga =
      (unsigned long long)(const void*)(Wt + (size_t)n0 * HD);
  unsigned galo = (unsigned)__builtin_amdgcn_readfirstlane((int)(ga & 0xffffffffu));
  unsigned gahi = (unsigned)__builtin_amdgcn_readfirstlane((int)(ga >> 32));
  unsigned lo   = (unsigned)__builtin_amdgcn_readfirstlane((int)ldsoff);

  u32x4 g0 = { 1u,                                   // count=1 (valid user D#)
               lo,                                   // lds_addr (bytes)
               galo,                                 // global_addr[31:0]
               (gahi & 0x01ffffffu) | (2u << 30) };  // global_addr[56:32]|type=2

  const unsigned td0 = (unsigned)NV * (unsigned)HD;  // tensor_dim0 (elements)
  i32x8 g1 = { (int)(1u << 16),                      // data_size=1 -> 2 bytes/elem
               (int)((td0 & 0xffffu) << 16),         // tensor_dim0[15:0]
               (int)((td0 >> 16) & 0xffffu),         // tensor_dim0[31:16]
               (int)(4096u << 16),                   // tile_dim0 = 4096 elements
               0,                                    // tile_dim1=0, tile_dim2=0
               (int)td0,                             // tensor_dim0_stride (lo)
               0, 0 };
  i32x4 gz = { 0, 0, 0, 0 };
#if defined(__clang_major__) && (__clang_major__ >= 23)
  i32x8 gz8 = { 0, 0, 0, 0, 0, 0, 0, 0 };
  __builtin_amdgcn_tensor_load_to_lds(g0, g1, gz, gz, gz8, 0);
#else
  __builtin_amdgcn_tensor_load_to_lds(g0, g1, gz, gz, 0);
#endif
}

// ---------------- precompute: W_out2 (f32 [HD][NV]) -> Wt (bf16 [NV][HD]) ----------------
__global__ __launch_bounds__(256) void k_wt(const float* __restrict__ W2,
                                            unsigned short* __restrict__ Wt) {
  int v = blockIdx.x;           // 0..NV-1
  int h = threadIdx.x;          // 0..255
  float f = W2[(size_t)h * NV + v];
  __bf16 b = (__bf16)f;
  Wt[(size_t)v * HD + h] = __builtin_bit_cast(unsigned short, b);
}

// ---------------- precompute: enc_feats = enc_out @ W_enc, per batch ----------------
__global__ __launch_bounds__(256) void k_encf(const float* __restrict__ enc_out,
                                              const float* __restrict__ W_enc,
                                              float* __restrict__ encf) {
  int b = blockIdx.x;
  int d = threadIdx.x;          // output feature
  const float* eb = enc_out + (size_t)b * SLEN * HD;
  float* ob = encf + (size_t)b * SLEN * HD;
  for (int s = 0; s < SLEN; ++s) {
    float acc = 0.f;
    const float* row = eb + (size_t)s * HD;
    for (int h = 0; h < HD; ++h) acc += row[h] * W_enc[h * HD + d];
    ob[(size_t)s * HD + d] = acc;
  }
}

// ---------------- main decode kernel: one workgroup per batch ----------------
__global__ __launch_bounds__(256)
void k_decode(const float* __restrict__ enc_out,
              const unsigned char* __restrict__ enc_mask,
              const int* __restrict__ enc_ids,
              const float* __restrict__ dih, const float* __restrict__ dic,
              const float* __restrict__ embedding,
              const float* __restrict__ W_in, const float* __restrict__ b_in,
              const float* __restrict__ W_lstm, const float* __restrict__ b_lstm,
              const float* __restrict__ W_dec, const float* __restrict__ b_attn,
              const float* __restrict__ v_attn,
              const float* __restrict__ w_pgen, const float* __restrict__ b_pgen,
              const float* __restrict__ W_out1, const float* __restrict__ b_out1,
              const float* __restrict__ b_out2,
              const unsigned short* __restrict__ Wt,
              const float* __restrict__ encf_g,
              float* __restrict__ dist_g,
              int* __restrict__ out_tok, float* __restrict__ out_lp)
{
  extern __shared__ char smraw[];
  Smem& sm = *reinterpret_cast<Smem*>(smraw);

  const int b    = blockIdx.x;
  const int tid  = threadIdx.x;
  const int lane = tid & 31;
  const int wv   = tid >> 5;

  const float* encf  = encf_g + (size_t)b * SLEN * HD;
  const float* eoutB = enc_out + (size_t)b * SLEN * HD;
  const unsigned char* maskB = enc_mask + (size_t)b * SLEN;
  const int* idsB = enc_ids + (size_t)b * SLEN;
  float* dist = dist_g + (size_t)b * NBEAM * NVE;

  // ---- init state ----
  for (int i = tid; i < NBEAM * HD; i += 256) {
    int hh = i % HD;
    sm.h[i] = dih[b * HD + hh];
    sm.c[i] = dic[b * HD + hh];
    sm.ctx[i] = 0.f;
  }
  for (int i = tid; i < HD; i += 256) { sm.vatt[i] = v_attn[i]; sm.batt[i] = b_attn[i]; }
  for (int i = tid; i < NBEAM * NSTEP; i += 256) sm.tokens[i] = 0;
  if (tid < NBEAM) { sm.lp[tid] = (tid == 0) ? 0.f : -1e9f; sm.tokc[tid] = 2; /*START*/ }
  __syncthreads();

  for (int t = 0; t < NSTEP; ++t) {
    // ---- Ph1: embed previous tokens (OOV -> UNK=0) ----
    for (int i = tid; i < NBEAM * ED; i += 256) {
      int k = i / ED, e = i % ED;
      int tk = sm.tokc[k]; if (tk >= NV) tk = 0;
      sm.emb[i] = embedding[(size_t)tk * ED + e];
    }
    __syncthreads();

    // ---- Ph2: x = [emb|ctx] @ W_in + b_in ----
    for (int i = tid; i < NBEAM * ED; i += 256) {
      int k = i / ED, e = i % ED;
      float acc = b_in[e];
      const float* ek = sm.emb + k * ED;
      const float* ck = sm.ctx + k * HD;
      for (int j = 0; j < ED; ++j) acc += ek[j] * W_in[j * ED + e];
      for (int j = 0; j < HD; ++j) acc += ck[j] * W_in[(ED + j) * ED + e];
      sm.xbuf[i] = acc;
    }
    __syncthreads();

    // ---- Ph3: z = [x|h] @ W_lstm + b_lstm ----
    for (int i = tid; i < NBEAM * 4 * HD; i += 256) {
      int k = i / (4 * HD), q = i % (4 * HD);
      float acc = b_lstm[q];
      const float* xk = sm.xbuf + k * ED;
      const float* hk = sm.h + k * HD;
      for (int j = 0; j < ED; ++j) acc += xk[j] * W_lstm[j * (4 * HD) + q];
      for (int j = 0; j < HD; ++j) acc += hk[j] * W_lstm[(ED + j) * (4 * HD) + q];
      sm.z[i] = acc;
    }
    __syncthreads();

    // ---- Ph4: LSTM cell ----
    for (int i = tid; i < NBEAM * HD; i += 256) {
      int k = i / HD, hh = i % HD;
      float ig = sm.z[k * 4 * HD + 0 * HD + hh];
      float fg = sm.z[k * 4 * HD + 1 * HD + hh];
      float gg = sm.z[k * 4 * HD + 2 * HD + hh];
      float og = sm.z[k * 4 * HD + 3 * HD + hh];
      float cn = sigm(fg) * sm.c[i] + sigm(ig) * tanhf(gg);
      float hn = sigm(og) * tanhf(cn);
      sm.cN[i] = cn; sm.hN[i] = hn;
    }
    __syncthreads();

    // ---- Ph5: dec_feat = [c_n|h_n] @ W_dec ----
    for (int i = tid; i < NBEAM * HD; i += 256) {
      int k = i / HD, d = i % HD;
      float acc = 0.f;
      const float* ck = sm.cN + k * HD;
      const float* hk = sm.hN + k * HD;
      for (int j = 0; j < HD; ++j) acc += ck[j] * W_dec[j * HD + d];
      for (int j = 0; j < HD; ++j) acc += hk[j] * W_dec[(HD + j) * HD + d];
      sm.df[i] = acc;
    }
    __syncthreads();

    // ---- Ph6: attention scores, one wave per (k,s) task ----
    for (int task = wv; task < NBEAM * SLEN; task += 8) {
      int k = task / SLEN, s = task % SLEN;
      const float* ef  = encf + (size_t)s * HD;
      const float* dfk = sm.df + k * HD;
      float p = 0.f;
      int h0 = lane * 8;
      #pragma unroll
      for (int i2 = 0; i2 < 8; ++i2) {
        int hh = h0 + i2;
        p += sm.vatt[hh] * tanhf(ef[hh] + dfk[hh] + sm.batt[hh]);
      }
      p = wredsum(p);
      if (lane == 0) sm.att[k * SLEN + s] = maskB[s] ? p : -1e9f;
    }
    __syncthreads();

    // ---- Ph7: softmax over s per beam ----
    for (int k = 0; k < NBEAM; ++k) {
      float lm = -3.4e38f;
      for (int s = tid; s < SLEN; s += 256) lm = fmaxf(lm, sm.att[k * SLEN + s]);
      sm.red[tid] = lm; __syncthreads();
      for (int st = 128; st; st >>= 1) {
        if (tid < st) sm.red[tid] = fmaxf(sm.red[tid], sm.red[tid + st]);
        __syncthreads();
      }
      float mk = sm.red[0]; __syncthreads();
      float ls = 0.f;
      for (int s = tid; s < SLEN; s += 256) {
        float e2 = __expf(sm.att[k * SLEN + s] - mk);
        sm.att[k * SLEN + s] = e2; ls += e2;
      }
      sm.red[tid] = ls; __syncthreads();
      for (int st = 128; st; st >>= 1) {
        if (tid < st) sm.red[tid] += sm.red[tid + st];
        __syncthreads();
      }
      float inv = 1.f / sm.red[0]; __syncthreads();
      for (int s = tid; s < SLEN; s += 256) sm.att[k * SLEN + s] *= inv;
      __syncthreads();
    }

    // ---- Ph8: ctx_n = attn @ enc_outputs (thread = feature, 4 beam accumulators) ----
    {
      float a0 = 0, a1 = 0, a2 = 0, a3 = 0;
      int hh = tid;
      for (int s = 0; s < SLEN; ++s) {
        float e2 = eoutB[(size_t)s * HD + hh];
        a0 += sm.att[0 * SLEN + s] * e2;
        a1 += sm.att[1 * SLEN + s] * e2;
        a2 += sm.att[2 * SLEN + s] * e2;
        a3 += sm.att[3 * SLEN + s] * e2;
      }
      sm.ctxN[0 * HD + hh] = a0; sm.ctxN[1 * HD + hh] = a1;
      sm.ctxN[2 * HD + hh] = a2; sm.ctxN[3 * HD + hh] = a3;
    }
    __syncthreads();

    // ---- Ph9: p_gen (wave k computes one beam) ----
    if (wv < NBEAM) {
      int k = wv;
      float p = 0.f;
      for (int j = lane; j < 3 * HD + ED; j += 32) {
        float val;
        if      (j <     HD) val = sm.ctxN[k * HD + j];
        else if (j < 2 * HD) val = sm.cN[k * HD + j - HD];
        else if (j < 3 * HD) val = sm.hN[k * HD + j - 2 * HD];
        else                 val = sm.xbuf[k * ED + j - 3 * HD];
        p += val * w_pgen[j];
      }
      p = wredsum(p);
      if (lane == 0) sm.pg[k] = sigm(p + b_pgen[0]);
    }
    __syncthreads();

    // ---- Ph10: out1 = [h_n|ctx_n] @ W_out1 + b_out1 ; build padded bf16 A ----
    for (int i = tid; i < NBEAM * HD; i += 256) {
      int k = i / HD, d = i % HD;
      float acc = b_out1[d];
      const float* hk = sm.hN + k * HD;
      const float* ck = sm.ctxN + k * HD;
      for (int j = 0; j < HD; ++j) acc += hk[j] * W_out1[j * HD + d];
      for (int j = 0; j < HD; ++j) acc += ck[j] * W_out1[(HD + j) * HD + d];
      sm.out1[i] = acc;
    }
    __syncthreads();
    for (int i = tid; i < 16 * HD; i += 256) {
      int m = i / HD;
      float vf = (m < NBEAM) ? sm.out1[m * HD + (i % HD)] : 0.f;
      __bf16 bb = (__bf16)vf;
      sm.Apad[i] = __builtin_bit_cast(unsigned short, bb);
    }
    __syncthreads();

    // ---- Ph11: WMMA vocab projection: (16 x 256) @ (256 x 50000), bf16 -> f32
    //      B panels staged into LDS with the Tensor Data Mover (double buffer).
    {
      float lmax0 = -3.4e38f, lmax1 = -3.4e38f, lmax2 = -3.4e38f, lmax3 = -3.4e38f;
      const __bf16* A = (const __bf16*)sm.Apad;
      int m16 = lane & 15, kh = lane >> 4;

      unsigned short* stage = sm.bstage + (size_t)wv * 2 * PANEL;
      unsigned soff[2];
      soff[0] = (unsigned)(size_t)(void*)(stage);
      soff[1] = (unsigned)(size_t)(void*)(stage + PANEL);

      // prologue: stage this wave's first tile
      tdm_load_panel(Wt, wv * 16, soff[0]);

      int buf = 0;
      for (int tile = wv; tile < NTIL; tile += 8) {
        int nxt = tile + 8;
        if (nxt < NTIL) {
          tdm_load_panel(Wt, nxt * 16, soff[buf ^ 1]);
          __builtin_amdgcn_s_wait_tensorcnt(1);   // current panel landed
        } else {
          __builtin_amdgcn_s_wait_tensorcnt(0);
        }
        const __bf16* Bp = (const __bf16*)(stage + buf * PANEL);

        int n0 = tile * 16;
        v8f acc = {};
        #pragma unroll
        for (int kb = 0; kb < HD; kb += 32) {
          // A fragment: lane(m=lane&15, kh=lane>>4): K = kb+kh*16..+15 contiguous
          v16bf af;
          const __bf16* arow = A + m16 * HD + kb + kh * 16;
          #pragma unroll
          for (int i2 = 0; i2 < 8; ++i2) { af[i2] = arow[i2]; af[8 + i2] = arow[8 + i2]; }
          // B fragment from LDS stage: lane holds column n0+(lane&15), K contiguous
          const v16bf* bp = (const v16bf*)(Bp + (size_t)m16 * HD + kb + kh * 16);
          v16bf bf_ = *bp;
          acc = __builtin_amdgcn_wmma_f32_16x16x32_bf16(
              false, af, false, bf_, (short)0, acc, false, false);
        }
        if (lane < 16) {
          int n = n0 + lane;
          float bo = b_out2[n];
          float l0 = acc[0] + bo, l1 = acc[1] + bo, l2 = acc[2] + bo, l3 = acc[3] + bo;
          dist[0 * NVE + n] = l0; dist[1 * NVE + n] = l1;
          dist[2 * NVE + n] = l2; dist[3 * NVE + n] = l3;
          lmax0 = fmaxf(lmax0, l0); lmax1 = fmaxf(lmax1, l1);
          lmax2 = fmaxf(lmax2, l2); lmax3 = fmaxf(lmax3, l3);
        }
        buf ^= 1;
      }
      lmax0 = wredmax(lmax0); lmax1 = wredmax(lmax1);
      lmax2 = wredmax(lmax2); lmax3 = wredmax(lmax3);
      if (lane == 0) {
        sm.wavemax[wv * NBEAM + 0] = lmax0; sm.wavemax[wv * NBEAM + 1] = lmax1;
        sm.wavemax[wv * NBEAM + 2] = lmax2; sm.wavemax[wv * NBEAM + 3] = lmax3;
      }
    }
    __threadfence();
    __syncthreads();
    if (tid < NBEAM) {
      float m2 = -3.4e38f;
      for (int w2 = 0; w2 < 8; ++w2) m2 = fmaxf(m2, sm.wavemax[w2 * NBEAM + tid]);
      sm.mx[tid] = m2;
    }
    __syncthreads();

    // ---- Ph12: vocab softmax (exp + sum) ----
    {
      float sl[4] = {0.f, 0.f, 0.f, 0.f};
      float m0 = sm.mx[0], m1 = sm.mx[1], m2 = sm.mx[2], m3 = sm.mx[3];
      for (int n = tid; n < NV; n += 256) {
        float e0 = __expf(dist[0 * NVE + n] - m0); dist[0 * NVE + n] = e0; sl[0] += e0;
        float e1 = __expf(dist[1 * NVE + n] - m1); dist[1 * NVE + n] = e1; sl[1] += e1;
        float e2 = __expf(dist[2 * NVE + n] - m2); dist[2 * NVE + n] = e2; sl[2] += e2;
        float e3 = __expf(dist[3 * NVE + n] - m3); dist[3 * NVE + n] = e3; sl[3] += e3;
      }
      for (int kk = 0; kk < NBEAM; ++kk) {
        sm.red[tid] = sl[kk]; __syncthreads();
        for (int st = 128; st; st >>= 1) {
          if (tid < st) sm.red[tid] += sm.red[tid + st];
          __syncthreads();
        }
        if (tid == 0) sm.sums[kk] = sm.red[0];
        __syncthreads();
      }
    }

    // ---- Ph13: scale by p_gen/sum, zero OOV tail ----
    {
      float c0 = sm.pg[0] / sm.sums[0], c1 = sm.pg[1] / sm.sums[1];
      float c2 = sm.pg[2] / sm.sums[2], c3 = sm.pg[3] / sm.sums[3];
      for (int n = tid; n < NV; n += 256) {
        dist[0 * NVE + n] *= c0; dist[1 * NVE + n] *= c1;
        dist[2 * NVE + n] *= c2; dist[3 * NVE + n] *= c3;
      }
      for (int n = NV + tid; n < NVE; n += 256) {
        dist[0 * NVE + n] = 0.f; dist[1 * NVE + n] = 0.f;
        dist[2 * NVE + n] = 0.f; dist[3 * NVE + n] = 0.f;
      }
    }
    __threadfence();
    __syncthreads();

    // ---- Ph14: pointer scatter: dist[k][ids[s]] += (1-p_gen)*attn ----
    for (int i2 = tid; i2 < NBEAM * SLEN; i2 += 256) {
      int k = i2 / SLEN, s = i2 % SLEN;
      float w = (1.f - sm.pg[k]) * sm.att[k * SLEN + s];
      atomicAdd(&dist[(size_t)k * NVE + idsB[s]], w);
    }
    __threadfence();
    __syncthreads();

    // ---- Ph15: top-4 over logp + log(final + 1e-10), idx = k*NVE + n ----
    {
      float tv[4] = {-3.4e38f, -3.4e38f, -3.4e38f, -3.4e38f};
      int   ti[4] = {0, 0, 0, 0};
      for (int idx = tid; idx < NBEAM * NVE; idx += 256) {
        int k = idx / NVE;
        float val = sm.lp[k] + __logf(dist[idx] + 1e-10f);
        if (val > tv[3]) {
          int p = 3;
          while (p > 0 && tv[p - 1] < val) { tv[p] = tv[p - 1]; ti[p] = ti[p - 1]; --p; }
          tv[p] = val; ti[p] = idx;
        }
      }
      #pragma unroll
      for (int j = 0; j < 4; ++j) { sm.tkv[tid * 4 + j] = tv[j]; sm.tki[tid * 4 + j] = ti[j]; }
      __syncthreads();
      for (int st = 128; st; st >>= 1) {
        if (tid < st) {
          float* tvp = &sm.tkv[tid * 4]; int* tip = &sm.tki[tid * 4];
          #pragma unroll
          for (int j = 0; j < 4; ++j) {
            float v2 = sm.tkv[(tid + st) * 4 + j];
            int  id2 = sm.tki[(tid + st) * 4 + j];
            if (v2 > tvp[3]) {
              int p = 3;
              while (p > 0 && tvp[p - 1] < v2) { tvp[p] = tvp[p - 1]; tip[p] = tip[p - 1]; --p; }
              tvp[p] = v2; tip[p] = id2;
            }
          }
        }
        __syncthreads();
      }
      if (tid == 0) {
        for (int k = 0; k < NBEAM; ++k) {
          int idx = sm.tki[k];
          sm.vals[k] = sm.tkv[k];
          sm.bsel[k] = idx / NVE;
          sm.ntok[k] = idx % NVE;
        }
      }
    }
    __syncthreads();

    // ---- Ph16: beam reorder + token append ----
    for (int i2 = tid; i2 < NBEAM * HD; i2 += 256) {
      int k = i2 / HD, hh = i2 % HD, bs = sm.bsel[k];
      sm.h[i2]   = sm.hN[bs * HD + hh];
      sm.c[i2]   = sm.cN[bs * HD + hh];
      sm.ctx[i2] = sm.ctxN[bs * HD + hh];
    }
    for (int i2 = tid; i2 < NBEAM * NSTEP; i2 += 256) {
      int k = i2 / NSTEP, j = i2 % NSTEP;
      sm.tokensN[i2] = sm.tokens[sm.bsel[k] * NSTEP + j];
    }
    __syncthreads();
    for (int i2 = tid; i2 < NBEAM * NSTEP; i2 += 256) {
      int k = i2 / NSTEP, j = i2 % NSTEP;
      sm.tokens[i2] = (j == t) ? sm.ntok[k] : sm.tokensN[i2];
    }
    if (tid < NBEAM) { sm.lp[tid] = sm.vals[tid]; sm.tokc[tid] = sm.ntok[tid]; }
    __syncthreads();
  } // steps

  // ---- outputs: tokens (int32) then logp (f32), concatenated flat ----
  for (int i = tid; i < NBEAM * NSTEP; i += 256)
    out_tok[b * NBEAM * NSTEP + i] = sm.tokens[i];
  if (tid < NBEAM) out_lp[b * NBEAM + tid] = sm.lp[tid];
}

// ---------------- host launcher ----------------
extern "C" void kernel_launch(void* const* d_in, const int* in_sizes, int n_in,
                              void* d_out, int out_size, void* d_ws, size_t ws_size,
                              hipStream_t stream) {
  const float* enc_outputs = (const float*)d_in[0];
  const unsigned char* enc_mask = (const unsigned char*)d_in[1];
  const int*   enc_ids  = (const int*)d_in[2];
  const float* dih      = (const float*)d_in[3];
  const float* dic      = (const float*)d_in[4];
  const float* embedding= (const float*)d_in[5];
  const float* W_in     = (const float*)d_in[6];
  const float* b_in     = (const float*)d_in[7];
  const float* W_lstm   = (const float*)d_in[8];
  const float* b_lstm   = (const float*)d_in[9];
  const float* W_enc    = (const float*)d_in[10];
  const float* W_dec    = (const float*)d_in[11];
  const float* b_attn   = (const float*)d_in[12];
  const float* v_attn   = (const float*)d_in[13];
  const float* w_pgen   = (const float*)d_in[14];
  const float* b_pgen   = (const float*)d_in[15];
  const float* W_out1   = (const float*)d_in[16];
  const float* b_out1   = (const float*)d_in[17];
  const float* W_out2   = (const float*)d_in[18];
  const float* b_out2   = (const float*)d_in[19];

  // workspace carve: Wt(bf16) | enc_feats(f32) | extended distributions(f32)
  char* ws = (char*)d_ws;
  size_t off = 0;
  unsigned short* Wt = (unsigned short*)(ws + off);
  off += (size_t)NV * HD * sizeof(unsigned short);
  off = (off + 255) & ~(size_t)255;
  float* encf = (float*)(ws + off);
  off += (size_t)NB * SLEN * HD * sizeof(float);
  off = (off + 255) & ~(size_t)255;
  float* dist = (float*)(ws + off);
  off += (size_t)NB * NBEAM * NVE * sizeof(float);
  (void)off; (void)ws_size; (void)in_sizes; (void)n_in; (void)out_size;

  k_wt  <<<NV, 256, 0, stream>>>(W_out2, Wt);
  k_encf<<<NB, 256, 0, stream>>>(enc_outputs, W_enc, encf);

  int*   out_tok = (int*)d_out;
  float* out_lp  = (float*)d_out + (size_t)NB * NBEAM * NSTEP;

  size_t smem = sizeof(Smem);
  k_decode<<<NB, 256, smem, stream>>>(
      enc_outputs, enc_mask, enc_ids, dih, dic, embedding,
      W_in, b_in, W_lstm, b_lstm, W_dec, b_attn, v_attn,
      w_pgen, b_pgen, W_out1, b_out1, b_out2,
      Wt, encf, dist, out_tok, out_lp);
}
